// CustomizedRNN_13503377178895
// MI455X (gfx1250) — matrix operand order repaired and checked by reading, not compiled, not run on hardware
//
#include <hip/hip_runtime.h>
#include <hip/hip_bf16.h>
#include <stdint.h>
#include <math.h>

// ---- problem sizes (from reference) ----
constexpr int B_  = 64;
constexpr int T_  = 512;
constexpr int E_  = 300;
constexpr int EP_ = 320;   // E padded to multiple of 32 for WMMA K loop
constexpr int H1_ = 256;
constexpr int H2_ = 512;
constexpr int D1_ = 128;
constexpr int D2_ = 64;

typedef __attribute__((ext_vector_type(16))) __bf16 v16bf;
typedef __attribute__((ext_vector_type(8)))  __bf16 v8bf;
typedef __attribute__((ext_vector_type(8)))  float  v8f;

// Pin load-batch-then-WMMA-chain schedules (prevents the allocator from
// sinking fragment loads next to each WMMA and serializing on one reg pair).
#define SCHED_FENCE() __builtin_amdgcn_sched_barrier(0)

// ---------------------------------------------------------------------------
// WMMA + fragment helpers (CDNA5 ISA 7.12.2 layouts, wave32)
// ---------------------------------------------------------------------------
__device__ static inline v8f wmma_bf16(v16bf a, v16bf b, v8f c) {
  return __builtin_amdgcn_wmma_f32_16x16x32_bf16(
      false, a, false, b, (short)0, c, false, false);
}

__device__ static inline v16bf cat8(v8bf lo, v8bf hi) {
  return __builtin_shufflevector(lo, hi, 0, 1, 2, 3, 4, 5, 6, 7,
                                 8, 9, 10, 11, 12, 13, 14, 15);
}

// A tile 16x32 (MxK) bf16, row-major with 16B-aligned rows:
// per lane = two contiguous 16B runs -> 2x b128 loads.
__device__ static inline v16bf load_a_frag(const __bf16* tile, int pitch, int lane) {
  int m  = lane & 15;
  int hi = (lane >> 4) & 1;
  const __bf16* p = tile + m * pitch + hi * 8;
  return cat8(*(const v8bf*)p, *(const v8bf*)(p + 16));
}

// B fragment from frag-swizzled weights: 32B contiguous per lane -> 2x b128.
__device__ static inline v16bf load_swz_frag(const __bf16* fragbase, int lane) {
  const __bf16* p = fragbase + lane * 16;
  return cat8(*(const v8bf*)p, *(const v8bf*)(p + 8));
}

__device__ static inline v8bf cvt8_bf(v8f x) {
  v8bf r;
#pragma unroll
  for (int i = 0; i < 8; ++i) r[i] = (__bf16)x[i];
  return r;
}

__device__ static inline v8f cvt8_f32(v8bf x) {
  v8f r;
#pragma unroll
  for (int i = 0; i < 8; ++i) r[i] = (float)x[i];
  return r;
}

// ---------------------------------------------------------------------------
// Weight conversion f32 [K][N] -> bf16 frag-swizzled layout (KC = K/32 chunks)
// ---------------------------------------------------------------------------
template <int KC>
__global__ void k_cvt_swz(const float* __restrict__ src, __bf16* __restrict__ dst,
                          int Kvalid, int N) {
  int i = blockIdx.x * blockDim.x + threadIdx.x;
  if (i >= KC * 32 * N) return;
  int j    = i & 15;
  int lane = (i >> 4) & 31;
  int rest = i >> 9;
  int c    = rest % KC;
  int nt   = rest / KC;
  int n = nt * 16 + (lane & 15);
  int k = c * 32 + ((lane >> 4) & 1) * 8 + (j & 7) + ((j >> 3) << 4);
  float v = (k < Kvalid) ? src[k * N + n] : 0.f;
  dst[i] = (__bf16)v;
}

// ---------------------------------------------------------------------------
// xw1 = emb[tokens] @ W1x + b1  -> blocked C-layout bf16
// ---------------------------------------------------------------------------
__global__ __launch_bounds__(256) void k_xw1(
    const int* __restrict__ tokens, const float* __restrict__ emb,
    const __bf16* __restrict__ w1x, const float* __restrict__ b1,
    __bf16* __restrict__ xw1) {
  __shared__ __align__(16) __bf16 As[64][EP_];   // 40 KB
  int t = blockIdx.x;
  int tid = threadIdx.x, lane = tid & 31, wave = tid >> 5;
  int mt = wave & 3;
  int ng = wave >> 2;

  {
    int row = tid >> 2, tp = tid & 3;
    int tok = tokens[row * T_ + t];
    const float* er = emb + (size_t)tok * E_;
#pragma unroll 8
    for (int i = 0; i < EP_ / 4; ++i) {
      int k = i * 4 + tp;
      As[row][k] = (k < E_) ? (__bf16)er[k] : (__bf16)0.f;
    }
  }

  v8f acc[8];
  int ncol0 = ng * 128 + (lane & 15);
#pragma unroll
  for (int nt = 0; nt < 8; ++nt) {
    float bv = b1[ncol0 + nt * 16];
#pragma unroll
    for (int r = 0; r < 8; ++r) acc[nt][r] = bv;
  }

  __syncthreads();

#pragma unroll
  for (int c = 0; c < EP_ / 32; ++c) {
    v16bf af = load_a_frag(&As[mt * 16][c * 32], EP_, lane);
    v16bf bfr[8];
#pragma unroll
    for (int nt = 0; nt < 8; ++nt)
      bfr[nt] = load_swz_frag(w1x + (size_t)((ng * 8 + nt) * 10 + c) * 512, lane);
    SCHED_FENCE();
#pragma unroll
    for (int nt = 0; nt < 8; ++nt)
      acc[nt] = wmma_bf16(af, bfr[nt], acc[nt]);
    SCHED_FENCE();
  }

#pragma unroll
  for (int nt = 0; nt < 8; ++nt) {
    __bf16* p = xw1 + (size_t)((t * 16 + ng * 8 + nt) * 4 + mt) * 256 + lane * 8;
    *(v8bf*)p = cvt8_bf(acc[nt]);
  }
}

// ---------------------------------------------------------------------------
// xw2 = seq1 @ W2x + b2  -> blocked C-layout bf16 (32 ntiles); no LDS.
// ---------------------------------------------------------------------------
__global__ __launch_bounds__(512) void k_xw2(
    const __bf16* __restrict__ seq1, const __bf16* __restrict__ w2x,
    const float* __restrict__ b2, __bf16* __restrict__ xw2) {
  int t = blockIdx.x;
  int tid = threadIdx.x, lane = tid & 31, wave = tid >> 5;
  int mt = wave & 3;
  int ng = wave >> 2;

  v8f acc[8];
  int ncol0 = ng * 128 + (lane & 15);
#pragma unroll
  for (int nt = 0; nt < 8; ++nt) {
    float bv = b2[ncol0 + nt * 16];
#pragma unroll
    for (int r = 0; r < 8; ++r) acc[nt][r] = bv;
  }

  const __bf16* arow = seq1 + (size_t)t * 64 * H1_ + (size_t)(mt * 16) * H1_;
#pragma unroll
  for (int c = 0; c < H1_ / 32; ++c) {
    v16bf af = load_a_frag(arow + c * 32, H1_, lane);
    v16bf bfr[8];
#pragma unroll
    for (int nt = 0; nt < 8; ++nt)
      bfr[nt] = load_swz_frag(w2x + (size_t)((ng * 8 + nt) * 8 + c) * 512, lane);
    SCHED_FENCE();
#pragma unroll
    for (int nt = 0; nt < 8; ++nt)
      acc[nt] = wmma_bf16(af, bfr[nt], acc[nt]);
    SCHED_FENCE();
  }

#pragma unroll
  for (int nt = 0; nt < 8; ++nt) {
    __bf16* p = xw2 + (size_t)((t * 32 + ng * 8 + nt) * 4 + mt) * 256 + lane * 8;
    *(v8bf*)p = cvt8_bf(acc[nt]);
  }
}

// ---------------------------------------------------------------------------
// Layer-1 scan: persistent single workgroup (16 waves).
// ---------------------------------------------------------------------------
__global__ __launch_bounds__(512) void k_rnn1(
    const __bf16* __restrict__ xw1, const __bf16* __restrict__ w1h,
    __bf16* __restrict__ seq1) {
  __shared__ __align__(16) __bf16 hbuf[2][64][H1_];   // 64 KB
  int tid = threadIdx.x, lane = tid & 31, wave = tid >> 5;  // wave = N tile

  v16bf wfrag[8];
#pragma unroll
  for (int c = 0; c < 8; ++c)
    wfrag[c] = load_swz_frag(w1h + (size_t)(wave * 8 + c) * 512, lane);

  {
    uint32_t* z = (uint32_t*)&hbuf[0][0][0];
    for (int i = tid; i < 2 * 64 * H1_ / 2; i += 512) z[i] = 0u;
  }
  __syncthreads();

  int nbase = wave * 16 + (lane & 15);
  int hi8 = (lane >> 4) << 3;

  for (int t = 0; t < T_; ++t) {
    const __bf16* cur = &hbuf[t & 1][0][0];
    __bf16* nxt = &hbuf[(t & 1) ^ 1][0][0];
    __bf16* srow = seq1 + (size_t)t * 64 * H1_;
    if (t + 1 < T_)
      __builtin_prefetch(xw1 + (size_t)(((t + 1) * 16 + wave) * 4) * 256 + lane * 8,
                         0, 3);
#pragma unroll
    for (int m = 0; m < 4; ++m) {
      v8f acc = cvt8_f32(
          *(const v8bf*)(xw1 + (size_t)((t * 16 + wave) * 4 + m) * 256 + lane * 8));
      v16bf afs[8];
#pragma unroll
      for (int c = 0; c < 8; ++c)
        afs[c] = load_a_frag(cur + m * 16 * H1_ + c * 32, H1_, lane);
      SCHED_FENCE();
#pragma unroll
      for (int c = 0; c < 8; ++c)
        acc = wmma_bf16(afs[c], wfrag[c], acc);
      SCHED_FENCE();
#pragma unroll
      for (int r = 0; r < 8; ++r) {
        float v = acc[r] > 0.f ? acc[r] : 0.f;
        __bf16 bv = (__bf16)v;
        nxt[(m * 16 + hi8 + r) * H1_ + nbase] = bv;
        srow[(m * 16 + hi8 + r) * H1_ + nbase] = bv;
      }
    }
    __syncthreads();
  }
}

// ---------------------------------------------------------------------------
// Layer-2 scan: persistent single workgroup (32 waves); W2h software-
// pipelined one K-chunk ahead out of L2.
// ---------------------------------------------------------------------------
__global__ __launch_bounds__(1024) void k_rnn2(
    const __bf16* __restrict__ xw2, const __bf16* __restrict__ w2h,
    float* __restrict__ hout) {
  __shared__ __align__(16) __bf16 hbuf[2][64][H2_];   // 128 KB
  int tid = threadIdx.x, lane = tid & 31, wave = tid >> 5;  // wave = N tile

  {
    uint32_t* z = (uint32_t*)&hbuf[0][0][0];
    for (int i = tid; i < 2 * 64 * H2_ / 2; i += 1024) z[i] = 0u;
  }
  __syncthreads();

  int nbase = wave * 16 + (lane & 15);
  int hi8 = (lane >> 4) << 3;
  const __bf16* wbase = w2h + (size_t)(wave * 16) * 512;

  for (int t = 0; t < T_; ++t) {
    const __bf16* cur = &hbuf[t & 1][0][0];
    __bf16* nxt = &hbuf[(t & 1) ^ 1][0][0];
    if (t + 1 < T_)
      __builtin_prefetch(xw2 + (size_t)(((t + 1) * 32 + wave) * 4) * 256 + lane * 8,
                         0, 3);

    v8f acc[4];
#pragma unroll
    for (int m = 0; m < 4; ++m)
      acc[m] = cvt8_f32(
          *(const v8bf*)(xw2 + (size_t)((t * 32 + wave) * 4 + m) * 256 + lane * 8));

    v16bf wf = load_swz_frag(wbase, lane);      // chunk 0
#pragma unroll 4
    for (int c = 0; c < 16; ++c) {              // weights L2-resident
      int cn = (c < 15) ? (c + 1) : c;
      v16bf wfn = load_swz_frag(wbase + (size_t)cn * 512, lane);  // lookahead
      v16bf a0 = load_a_frag(cur + 0 * 16 * H2_ + c * 32, H2_, lane);
      v16bf a1 = load_a_frag(cur + 1 * 16 * H2_ + c * 32, H2_, lane);
      v16bf a2 = load_a_frag(cur + 2 * 16 * H2_ + c * 32, H2_, lane);
      v16bf a3 = load_a_frag(cur + 3 * 16 * H2_ + c * 32, H2_, lane);
      SCHED_FENCE();
      acc[0] = wmma_bf16(a0, wf, acc[0]);
      acc[1] = wmma_bf16(a1, wf, acc[1]);
      acc[2] = wmma_bf16(a2, wf, acc[2]);
      acc[3] = wmma_bf16(a3, wf, acc[3]);
      SCHED_FENCE();
      wf = wfn;
    }

#pragma unroll
    for (int m = 0; m < 4; ++m)
#pragma unroll
      for (int r = 0; r < 8; ++r) {
        float v = acc[m][r] > 0.f ? acc[m][r] : 0.f;
        nxt[(m * 16 + hi8 + r) * H2_ + nbase] = (__bf16)v;
        if (t == T_ - 1) hout[(m * 16 + hi8 + r) * H2_ + nbase] = v;
      }
    __syncthreads();
  }
}

// ---------------------------------------------------------------------------
// MLP head: [64,512] -> relu 128 -> relu 64 -> sigmoid 1  (tiny, plain VALU)
// ---------------------------------------------------------------------------
__global__ __launch_bounds__(256) void k_head(
    const float* __restrict__ h2,
    const float* __restrict__ wd1, const float* __restrict__ bd1,
    const float* __restrict__ wd2, const float* __restrict__ bd2,
    const float* __restrict__ wc, const float* __restrict__ bc,
    float* __restrict__ out) {
  __shared__ float d1[64][D1_];
  __shared__ float d2[64][D2_];
  int tid = threadIdx.x;
  for (int idx = tid; idx < 64 * D1_; idx += 256) {
    int b = idx >> 7, j = idx & (D1_ - 1);
    float s = bd1[j];
    for (int k = 0; k < H2_; ++k) s += h2[b * H2_ + k] * wd1[k * D1_ + j];
    d1[b][j] = s > 0.f ? s : 0.f;
  }
  __syncthreads();
  for (int idx = tid; idx < 64 * D2_; idx += 256) {
    int b = idx >> 6, j = idx & (D2_ - 1);
    float s = bd2[j];
    for (int k = 0; k < D1_; ++k) s += d1[b][k] * wd2[k * D2_ + j];
    d2[b][j] = s > 0.f ? s : 0.f;
  }
  __syncthreads();
  if (tid < 64) {
    float s = bc[0];
    for (int k = 0; k < D2_; ++k) s += d2[tid][k] * wc[k];
    out[tid] = 1.f / (1.f + __expf(-s));
  }
}

// ---------------------------------------------------------------------------
// Launch
// ---------------------------------------------------------------------------
extern "C" void kernel_launch(void* const* d_in, const int* in_sizes, int n_in,
                              void* d_out, int out_size, void* d_ws, size_t ws_size,
                              hipStream_t stream) {
  const int*   tokens = (const int*)  d_in[0];
  const float* emb    = (const float*)d_in[1];
  const float* W1x    = (const float*)d_in[2];
  const float* W1h    = (const float*)d_in[3];
  const float* b1     = (const float*)d_in[4];
  const float* W2x    = (const float*)d_in[5];
  const float* W2h    = (const float*)d_in[6];
  const float* b2     = (const float*)d_in[7];
  const float* Wd1    = (const float*)d_in[8];
  const float* bd1    = (const float*)d_in[9];
  const float* Wd2    = (const float*)d_in[10];
  const float* bd2    = (const float*)d_in[11];
  const float* Wc     = (const float*)d_in[12];
  const float* bc     = (const float*)d_in[13];
  float* out = (float*)d_out;

  char* ws = (char*)d_ws;
  size_t o = 0;
  __bf16* w1x_sw  = (__bf16*)(ws + o); o += (size_t)EP_ * H1_ * 2;       // 160 KB
  __bf16* w1h_sw  = (__bf16*)(ws + o); o += (size_t)H1_ * H1_ * 2;       // 128 KB
  __bf16* w2x_sw  = (__bf16*)(ws + o); o += (size_t)H1_ * H2_ * 2;       // 256 KB
  __bf16* w2h_sw  = (__bf16*)(ws + o); o += (size_t)H2_ * H2_ * 2;       // 512 KB
  __bf16* xw1_bk  = (__bf16*)(ws + o); o += (size_t)T_ * B_ * H1_ * 2;   // 16 MB
  __bf16* seq1_bf = (__bf16*)(ws + o); o += (size_t)T_ * B_ * H1_ * 2;   // 16 MB
  __bf16* xw2_bk  = (__bf16*)(ws + o); o += (size_t)T_ * B_ * H2_ * 2;   // 32 MB
  float*  h2f     = (float*) (ws + o); o += (size_t)B_ * H2_ * 4;        // 128 KB

  // 1) weight conversion -> frag-swizzled bf16 (compile-time chunk counts)
  k_cvt_swz<10><<<(EP_ * H1_ + 255) / 256, 256, 0, stream>>>(W1x, w1x_sw, E_, H1_);
  k_cvt_swz<8><<<(H1_ * H1_ + 255) / 256, 256, 0, stream>>>(W1h, w1h_sw, H1_, H1_);
  k_cvt_swz<8><<<(H1_ * H2_ + 255) / 256, 256, 0, stream>>>(W2x, w2x_sw, H1_, H2_);
  k_cvt_swz<16><<<(H2_ * H2_ + 255) / 256, 256, 0, stream>>>(W2h, w2h_sw, H2_, H2_);

  // 2) fused embedding-gather + input projection (WMMA), bias folded
  k_xw1<<<T_, 256, 0, stream>>>(tokens, emb, w1x_sw, b1, xw1_bk);

  // 3) layer-1 recurrence (persistent, WMMA, weights in registers)
  k_rnn1<<<1, 512, 0, stream>>>(xw1_bk, w1h_sw, seq1_bf);

  // 4) layer-2 input projection (WMMA), bias folded
  k_xw2<<<T_, 512, 0, stream>>>(seq1_bf, w2x_sw, b2, xw2_bk);

  // 5) layer-2 recurrence (persistent, WMMA, weights streamed from L2)
  k_rnn2<<<1, 1024, 0, stream>>>(xw2_bk, w2h_sw, h2f);

  // 6) MLP head + sigmoid
  k_head<<<1, 256, 0, stream>>>(h2f, Wd1, bd1, Wd2, bd2, Wc, bc, out);
}
